// SelfAttention_80616536146761
// MI455X (gfx1250) — matrix operand order repaired and checked by reading, not compiled, not run on hardware
//
#include <hip/hip_runtime.h>
#include <hip/hip_bf16.h>

// Problem constants (reference: N=4, L=2048, E=1024, H=16, D=64)
#define N_B 4
#define L_L 2048
#define E_E 1024
#define H_H 16
#define D_D 64

typedef __attribute__((ext_vector_type(16))) __bf16 v16bf;
typedef __attribute__((ext_vector_type(8)))  __bf16 v8bf;
typedef __attribute__((ext_vector_type(8)))  float  v8f;
typedef __attribute__((ext_vector_type(4)))  unsigned int u32x4;
typedef __attribute__((ext_vector_type(8)))  int i32x8;
typedef __attribute__((ext_vector_type(4)))  int i32x4;

__device__ __forceinline__ v8f wmma_bf(v16bf a, v16bf b, v8f c) {
  // D = A(16x32 bf16) * B(32x16 bf16) + C(16x16 f32)
  return __builtin_amdgcn_wmma_f32_16x16x32_bf16(false, a, false, b, (short)0, c,
                                                 false, false);
}

// ---- TDM: 2-D tile load Global -> LDS (bf16 elements, data_size=2B) -----
// D# per CDNA5 ISA ch.8: group0 = {count, lds_addr, global_addr, type=2};
// group1 = {data_size, tensor_dim0/1, tile_dim0/1, tensor_dim0_stride}.
__device__ __forceinline__ void tdm_load_2d_bf16(
    unsigned lds_off, const void* gaddr, unsigned tensor_d0, unsigned tensor_d1,
    unsigned tile_d0, unsigned tile_d1, unsigned stride0_elems) {
  unsigned long long ga = (unsigned long long)(uintptr_t)gaddr;
  u32x4 g0;
  g0[0] = 1u;                                   // count=1 (valid), user mode
  g0[1] = lds_off;                              // LDS byte address
  g0[2] = (unsigned)ga;                         // global_addr[31:0]
  g0[3] = (unsigned)(ga >> 32) | (2u << 30);    // global_addr[56:32] | type=2
  i32x8 g1;
  g1[0] = (int)(1u << 16);                      // wg_mask=0, data_size=1 (2B)
  g1[1] = (int)(tensor_d0 << 16);               // barrier_addr=0 | td0[15:0]
  g1[2] = (int)((tensor_d0 >> 16) | (tensor_d1 << 16));
  g1[3] = (int)((tensor_d1 >> 16) | (tile_d0 << 16));
  g1[4] = (int)(tile_d1 & 0xFFFFu);             // tile_dim1 | tile_dim2=0
  g1[5] = (int)stride0_elems;                   // dim0_stride[31:0]
  g1[6] = 0;                                    // stride hi / dim1_stride (2D)
  g1[7] = 0;
  i32x4 z4 = {0, 0, 0, 0};                      // groups 2/3 unused (2-D)
  i32x8 z8 = {0, 0, 0, 0, 0, 0, 0, 0};
  __builtin_amdgcn_tensor_load_to_lds(g0, g1, z4, z4, z8, 0);
}

// ---- Fragment builders --------------------------------------------------
// A-matrix 16x32 bf16: lane holds row M=lane&15; elem i(0..15) ->
// K = kbase + 16*hi + i (contiguous, matching B).
template <typename T>
__device__ __forceinline__ v16bf load_a16_cvt(const T* rowptr, int kbase, int hi) {
  v16bf a;
  const T* p0 = rowptr + kbase + 16 * hi;
  const T* p1 = rowptr + kbase + 16 * hi + 8;
#pragma unroll
  for (int i = 0; i < 8; ++i) {
    a[i]     = (__bf16)p0[i];
    a[i + 8] = (__bf16)p1[i];
  }
  return a;
}

__device__ __forceinline__ v16bf load_a16_bf(const __bf16* rowptr, int kbase, int hi) {
  v8bf lo = *(const v8bf*)(rowptr + kbase + 16 * hi);
  v8bf hh = *(const v8bf*)(rowptr + kbase + 16 * hi + 8);
  return __builtin_shufflevector(lo, hh, 0, 1, 2, 3, 4, 5, 6, 7,
                                 8, 9, 10, 11, 12, 13, 14, 15);
}

// B-matrix 32x16 bf16: lane holds column N=lane&15; elem i -> K = 16*hi + i,
// i.e. 16 contiguous elements along K from a K-major row.
template <typename T>
__device__ __forceinline__ v16bf load_b16_cvt(const T* p) {
  v16bf b;
#pragma unroll
  for (int i = 0; i < 16; ++i) b[i] = (__bf16)p[i];
  return b;
}

__device__ __forceinline__ v16bf load_b16_bf(const __bf16* p) {
  return *(const v16bf*)p;  // 32B-aligned by construction
}

// ---- Kernel 0: Wo fp32 -> bf16 one-time conversion ----------------------
__global__ __launch_bounds__(256) void cvt_wo_kernel(const float* __restrict__ W,
                                                     __bf16* __restrict__ Wb) {
  const int idx = (blockIdx.x * 256 + threadIdx.x) * 4;
#pragma unroll
  for (int i = 0; i < 4; ++i) Wb[idx + i] = (__bf16)W[idx + i];
}

// ---- Kernel 1: per-head QKV projection ---------------------------------
// out[n,l,h,:] = W @ x[n,l,h,:]   (x @ W^T)
// transposed==0: write bf16 [nh][L][D] ; transposed==1: write bf16 [nh][D][L]
__global__ __launch_bounds__(256) void qkv_proj_kernel(
    const float* __restrict__ X, const float* __restrict__ W,
    __bf16* __restrict__ out, int transposed) {
  const int lane = threadIdx.x & 31;
  const int wv   = threadIdx.x >> 5;
  const int tile = blockIdx.x * 8 + wv;   // 8192 tiles = (N*H) * (L/16)
  const int lt   = tile & 127;
  const int nh   = tile >> 7;
  const int n = nh >> 4, h = nh & 15;
  const int col = lane & 15, hi = lane >> 4;

  const float* xrow = X + ((size_t)(n * L_L + lt * 16 + col) * E_E + h * D_D);
  v16bf a0 = load_a16_cvt(xrow, 0, hi);
  v16bf a1 = load_a16_cvt(xrow, 32, hi);

#pragma unroll
  for (int dt = 0; dt < 4; ++dt) {
    // B[k][d] = W[d][k] : lane col -> d = dt*16+col, contiguous k run.
    const float* wrow = W + (dt * 16 + col) * D_D;
    v8f acc = {};
    acc = wmma_bf(a0, load_b16_cvt(wrow + 16 * hi), acc);
    acc = wmma_bf(a1, load_b16_cvt(wrow + 32 + 16 * hi), acc);
    if (!transposed) {
#pragma unroll
      for (int r = 0; r < 8; ++r) {
        const int l = lt * 16 + r + 8 * hi;
        out[((size_t)nh * L_L + l) * D_D + dt * 16 + col] = (__bf16)acc[r];
      }
    } else {
#pragma unroll
      for (int r = 0; r < 8; ++r) {
        const int l = lt * 16 + r + 8 * hi;
        out[((size_t)nh * D_D + dt * 16 + col) * L_L + l] = (__bf16)acc[r];
      }
    }
  }
}

// ---- Kernel 2: flash attention ------------------------------------------
// One block = 8 waves, all sharing the same (n,h); wave w owns query rows
// [qt*16, qt*16+16). K/V tiles are staged in double-buffered LDS by the
// Tensor Data Mover: wave 0 issues tile kt+1 into buf^1 before waiting for
// tile kt (TDM ops complete in order -> s_wait_tensorcnt 2 == tile kt done).
__global__ __launch_bounds__(256) void flash_attn_kernel(
    const __bf16* __restrict__ Qb, const __bf16* __restrict__ Kb,
    const __bf16* __restrict__ Vt, const int* __restrict__ mask,
    __bf16* __restrict__ Ob) {
  __shared__ __attribute__((aligned(64))) __bf16 kTile[2][32 * D_D];  // [key][d]
  __shared__ __attribute__((aligned(64))) __bf16 vTile[2][D_D * 32];  // [d][l]
  __shared__ __attribute__((aligned(64))) __bf16 pbuf[8][16 * 32];    // per-wave

  const int lane = threadIdx.x & 31;
  const int wv   = threadIdx.x >> 5;
  const int nh   = blockIdx.x >> 4;            // 64 head-batches
  const int qt   = (blockIdx.x & 15) * 8 + wv; // 128 q-tiles per head
  const int n = nh >> 4, h = nh & 15;
  const int col = lane & 15, hi = lane >> 4;

  // softmax over energy/sqrt(E), computed in exp2 domain:
  const float kScale = 1.4426950408889634f / 32.0f;  // log2(e)/sqrt(1024)

  const __bf16* qrow = Qb + ((size_t)nh * L_L + qt * 16 + col) * D_D;
  v16bf aQ0 = load_a16_bf(qrow, 0, hi);
  v16bf aQ1 = load_a16_bf(qrow, 32, hi);

  v8f o0 = {}, o1 = {}, o2 = {}, o3 = {};
  float mrow[8], lrow[8];
#pragma unroll
  for (int r = 0; r < 8; ++r) { mrow[r] = -1e30f; lrow[r] = 0.0f; }

  const int*    mbase = mask + (size_t)n * L_L * L_L;
  const __bf16* kHead = Kb + (size_t)nh * L_L * D_D;
  const __bf16* vHead = Vt + (size_t)nh * D_D * L_L;
  const unsigned kOff0 = (unsigned)(uintptr_t)(void*)&kTile[0][0];
  const unsigned kOff1 = (unsigned)(uintptr_t)(void*)&kTile[1][0];
  const unsigned vOff0 = (unsigned)(uintptr_t)(void*)&vTile[0][0];
  const unsigned vOff1 = (unsigned)(uintptr_t)(void*)&vTile[1][0];

  if (wv == 0) {  // prologue: stage tile 0 into buffer 0
    tdm_load_2d_bf16(kOff0, kHead, D_D, L_L, D_D, 32, D_D);
    tdm_load_2d_bf16(vOff0, vHead, L_L, D_D, 32, D_D, L_L);
  }

  for (int kt = 0; kt < L_L / 32; ++kt) {
    const int kbase = kt * 32;
    const int buf   = kt & 1;
    if (wv == 0) {
      if (kt + 1 < L_L / 32) {  // issue next tile into the other buffer
        const int nb = kbase + 32;
        tdm_load_2d_bf16(buf ? kOff0 : kOff1, kHead + (size_t)nb * D_D,
                         D_D, L_L, D_D, 32, D_D);
        tdm_load_2d_bf16(buf ? vOff0 : vOff1, vHead + nb,
                         L_L, D_D, 32, D_D, L_L);
        __builtin_amdgcn_s_wait_tensorcnt(2);  // tile kt has landed
      } else {
        __builtin_amdgcn_s_wait_tensorcnt(0);
      }
    }
    __syncthreads();

    // ---- S = Q K^T (two 16-key column groups, K-dim = 64 in two chunks)
    const __bf16* k0 = &kTile[buf][0] + col * D_D;
    const __bf16* k1 = &kTile[buf][0] + (16 + col) * D_D;
    v8f s0 = {}, s1 = {};
    s0 = wmma_bf(aQ0, load_b16_bf(k0 + 16 * hi), s0);
    s0 = wmma_bf(aQ1, load_b16_bf(k0 + 32 + 16 * hi), s0);
    s1 = wmma_bf(aQ0, load_b16_bf(k1 + 16 * hi), s1);
    s1 = wmma_bf(aQ1, load_b16_bf(k1 + 32 + 16 * hi), s1);

    // ---- masked online softmax (row = r + 8*hi on vgpr index, col on lanes)
#pragma unroll
    for (int r = 0; r < 8; ++r) {
      const int qr = qt * 16 + r + 8 * hi;
      const int* mp = mbase + (size_t)qr * L_L + kbase;
      float v0 = mp[col]      ? s0[r] * kScale : -1e30f;
      float v1 = mp[16 + col] ? s1[r] * kScale : -1e30f;
      float mx = fmaxf(v0, v1);
      mx = fmaxf(mx, __shfl_xor(mx, 1, 16));
      mx = fmaxf(mx, __shfl_xor(mx, 2, 16));
      mx = fmaxf(mx, __shfl_xor(mx, 4, 16));
      mx = fmaxf(mx, __shfl_xor(mx, 8, 16));
      const float mnew  = fmaxf(mrow[r], mx);
      const float alpha = exp2f(mrow[r] - mnew);
      mrow[r] = mnew;
      const float p0 = exp2f(v0 - mnew);
      const float p1 = exp2f(v1 - mnew);
      float rs = p0 + p1;
      rs += __shfl_xor(rs, 1, 16);
      rs += __shfl_xor(rs, 2, 16);
      rs += __shfl_xor(rs, 4, 16);
      rs += __shfl_xor(rs, 8, 16);
      lrow[r] = lrow[r] * alpha + rs;
      o0[r] *= alpha; o1[r] *= alpha; o2[r] *= alpha; o3[r] *= alpha;
      pbuf[wv][(r + 8 * hi) * 32 + col]      = (__bf16)p0;
      pbuf[wv][(r + 8 * hi) * 32 + 16 + col] = (__bf16)p1;
    }
    // C-layout -> A-layout transpose through per-wave LDS (DS ops in-order;
    // wait for stores before the cross-lane re-read).
    asm volatile("s_wait_dscnt 0" ::: "memory");
    v16bf aP = load_a16_bf(&pbuf[wv][col * 32], 0, hi);

    // ---- O += P * V  (vTile[d][l]: 16 contiguous along l per lane)
    const __bf16* vb = &vTile[buf][0] + 16 * hi;
    o0 = wmma_bf(aP, load_b16_bf(vb + (0 * 16 + col) * 32), o0);
    o1 = wmma_bf(aP, load_b16_bf(vb + (1 * 16 + col) * 32), o1);
    o2 = wmma_bf(aP, load_b16_bf(vb + (2 * 16 + col) * 32), o2);
    o3 = wmma_bf(aP, load_b16_bf(vb + (3 * 16 + col) * 32), o3);

    __syncthreads();  // all waves done with buf before TDM refills it
  }

  // ---- finalize: divide by row sums, write bf16 [n][l][h*64+d]
#pragma unroll
  for (int r = 0; r < 8; ++r) {
    const float inv = 1.0f / lrow[r];
    const int l = qt * 16 + r + 8 * hi;
    __bf16* orow = Ob + ((size_t)(n * L_L + l) * E_E + h * D_D);
    orow[0  + col] = (__bf16)(o0[r] * inv);
    orow[16 + col] = (__bf16)(o1[r] * inv);
    orow[32 + col] = (__bf16)(o2[r] * inv);
    orow[48 + col] = (__bf16)(o3[r] * inv);
  }
}

// ---- Kernel 3: output projection GEMM  out = Ob @ Wo^T + bo -------------
// M=8192, N=1024, K=1024. 8 waves of a block share one 64-column e-group;
// the 64x32 Wo^T tile per K-chunk is TDM-staged into double-buffered LDS.
__global__ __launch_bounds__(256) void out_proj_kernel(
    const __bf16* __restrict__ Ob, const __bf16* __restrict__ WoB,
    const float* __restrict__ bo, float* __restrict__ out) {
  __shared__ __attribute__((aligned(64))) __bf16 wTile[2][64 * 32];  // [e][k]

  const int lane = threadIdx.x & 31;
  const int wv   = threadIdx.x >> 5;
  const int eg   = blockIdx.x >> 6;            // 16 e-groups of 64 cols
  const int mt   = (blockIdx.x & 63) * 8 + wv; // 512 m-tiles
  const int col = lane & 15, hi = lane >> 4;

  const __bf16* wBase = WoB + (size_t)(eg * 64) * E_E;
  const unsigned wOff0 = (unsigned)(uintptr_t)(void*)&wTile[0][0];
  const unsigned wOff1 = (unsigned)(uintptr_t)(void*)&wTile[1][0];

  if (wv == 0) {  // prologue: stage K-chunk 0
    tdm_load_2d_bf16(wOff0, wBase, E_E, 64, 32, 64, E_E);
  }

  const __bf16* arow = Ob + (size_t)(mt * 16 + col) * E_E;
  v8f acc0 = {}, acc1 = {}, acc2 = {}, acc3 = {};
  for (int kc = 0; kc < E_E / 32; ++kc) {
    const int buf = kc & 1;
    if (wv == 0) {
      if (kc + 1 < E_E / 32) {
        tdm_load_2d_bf16(buf ? wOff0 : wOff1, wBase + (kc + 1) * 32,
                         E_E, 64, 32, 64, E_E);
        __builtin_amdgcn_s_wait_tensorcnt(1);  // chunk kc has landed
      } else {
        __builtin_amdgcn_s_wait_tensorcnt(0);
      }
    }
    __syncthreads();

    v16bf a = load_a16_bf(arow, kc * 32, hi);
    const __bf16* w = &wTile[buf][0] + 16 * hi;
    acc0 = wmma_bf(a, load_b16_bf(w + (0 * 16 + col) * 32), acc0);
    acc1 = wmma_bf(a, load_b16_bf(w + (1 * 16 + col) * 32), acc1);
    acc2 = wmma_bf(a, load_b16_bf(w + (2 * 16 + col) * 32), acc2);
    acc3 = wmma_bf(a, load_b16_bf(w + (3 * 16 + col) * 32), acc3);

    __syncthreads();
  }
#pragma unroll
  for (int e = 0; e < 4; ++e) {
    const v8f acc = (e == 0) ? acc0 : (e == 1) ? acc1 : (e == 2) ? acc2 : acc3;
    const int ecol = eg * 64 + e * 16 + col;
    const float b = bo[ecol];
#pragma unroll
    for (int r = 0; r < 8; ++r) {
      out[(size_t)(mt * 16 + r + 8 * hi) * E_E + ecol] = acc[r] + b;
    }
  }
}

// ---- Host launch --------------------------------------------------------
extern "C" void kernel_launch(void* const* d_in, const int* in_sizes, int n_in,
                              void* d_out, int out_size, void* d_ws, size_t ws_size,
                              hipStream_t stream) {
  const float* values = (const float*)d_in[0];
  const float* keys   = (const float*)d_in[1];
  const float* query  = (const float*)d_in[2];
  const int*   mask   = (const int*)d_in[3];
  const float* Wv = (const float*)d_in[4];
  const float* Wk = (const float*)d_in[5];
  const float* Wq = (const float*)d_in[6];
  const float* Wo = (const float*)d_in[7];
  const float* bo = (const float*)d_in[8];
  float* out = (float*)d_out;
  (void)in_sizes; (void)n_in; (void)out_size; (void)ws_size;

  const size_t perMat = (size_t)N_B * H_H * L_L * D_D;  // 8,388,608 bf16
  __bf16* Qb  = (__bf16*)d_ws;        // [nh][L][D]
  __bf16* Kb  = Qb + perMat;          // [nh][L][D]
  __bf16* Vt  = Kb + perMat;          // [nh][D][L] (transposed)
  __bf16* Ob  = Vt + perMat;          // [n][L][E]
  __bf16* WoB = Ob + perMat;          // [E][E] bf16 weights

  dim3 blk(256);
  cvt_wo_kernel<<<E_E * E_E / 1024, blk, 0, stream>>>(Wo, WoB);
  qkv_proj_kernel<<<1024, blk, 0, stream>>>(query,  Wq, Qb, 0);
  qkv_proj_kernel<<<1024, blk, 0, stream>>>(keys,   Wk, Kb, 0);
  qkv_proj_kernel<<<1024, blk, 0, stream>>>(values, Wv, Vt, 1);
  flash_attn_kernel<<<1024, blk, 0, stream>>>(Qb, Kb, Vt, mask, Ob);
  out_proj_kernel<<<1024, blk, 0, stream>>>(Ob, WoB, bo, out);
}
